// Attention_40690520163106
// MI455X (gfx1250) — matrix operand order repaired and checked by reading, not compiled, not run on hardware
//
#include <hip/hip_runtime.h>
#include <math.h>

// ---------------------------------------------------------------------------
// Problem constants (from reference): B=2, S=4096, D=1024, H=16, HD=64, R=16
// ---------------------------------------------------------------------------
#define B_  2
#define S_  4096
#define D_  1024
#define H_  16
#define HD_ 64
#define R_  16
#define MT_ (B_ * S_)          // 8192 total rows

typedef __bf16 bf16;
typedef __attribute__((ext_vector_type(16))) __bf16 v16bf;
typedef __attribute__((ext_vector_type(8)))  float  v8f;
typedef __attribute__((ext_vector_type(4)))  unsigned u32x4;
typedef __attribute__((ext_vector_type(8)))  unsigned u32x8;

// ---- bf16 <-> f32 bit-level converts ---------------------------------------
__device__ __forceinline__ float bf2f(bf16 x) {
    unsigned short s = __builtin_bit_cast(unsigned short, x);
    unsigned u = ((unsigned)s) << 16;
    return __builtin_bit_cast(float, u);
}
__device__ __forceinline__ bf16 f2bf(float f) {
    unsigned u = __builtin_bit_cast(unsigned, f);
    u += 0x7FFFu + ((u >> 16) & 1u);            // round-to-nearest-even
    unsigned short s = (unsigned short)(u >> 16);
    return __builtin_bit_cast(bf16, s);
}

// ---- WMMA wrapper ----------------------------------------------------------
__device__ __forceinline__ v8f wmma_bf16(v16bf a, v16bf b, v8f c) {
    return __builtin_amdgcn_wmma_f32_16x16x32_bf16(
        false, a, false, b, (short)0, c, false, false);
}

// ---- CDNA5 async global -> LDS copy (ASYNCcnt path, ISA 08 §4) -------------
__device__ __forceinline__ void async_copy_b128(void* lds_dst, const void* gsrc) {
    unsigned lds = (unsigned)(size_t)lds_dst;
    unsigned long long g = (unsigned long long)(size_t)gsrc;
    asm volatile("global_load_async_to_lds_b128 %0, %1, off"
                 :: "v"(lds), "v"(g) : "memory");
}
__device__ __forceinline__ void wait_async() {
#if defined(__has_builtin)
#if __has_builtin(__builtin_amdgcn_s_wait_asynccnt)
    __builtin_amdgcn_s_wait_asynccnt(0);
    return;
#endif
#endif
    asm volatile("s_wait_asynccnt 0" ::: "memory");
}

// ---- CDNA5 Tensor Data Mover: one-instruction 2D tile DMA (ISA 08 §7-8) ----
// Loads a [rows x row_elems] bf16 tile (row stride row_stride_elems) from
// global into LDS, adding pad_amount DWORDs of LDS padding every
// pad_interval DWORDs (D# pad fields), so the LDS layout matches our padded
// tiles.  tensor dims == tile dims -> no OOB.  Issue from ONE wave only.
__device__ __forceinline__ void tdm_load_2d(void* lds_dst, const void* gsrc,
                                            unsigned row_elems, unsigned rows,
                                            unsigned long long row_stride_elems,
                                            unsigned pad_interval_code,
                                            unsigned pad_amount_code) {
    unsigned lds = (unsigned)(size_t)lds_dst;
    unsigned long long g = (unsigned long long)(size_t)gsrc;
    u32x4 g0;
    g0[0] = 1u;                                            // count=1 (user D#)
    g0[1] = lds;                                           // lds_addr
    g0[2] = (unsigned)(g & 0xFFFFFFFFu);                   // global_addr[31:0]
    g0[3] = (unsigned)((g >> 32) & 0x01FFFFFFu)            // global_addr[56:32]
          | (2u << 30);                                    // type=2 ("image")
    u32x8 g1;
    g1[0] = (1u << 16)                                     // data_size=1 -> 2B
          | (1u << 20)                                     // pad_enable
          | (pad_interval_code << 22)
          | (pad_amount_code << 25);
    g1[1] = (row_elems & 0xFFFFu) << 16;                   // tensor_dim0[15:0]
    g1[2] = ((row_elems >> 16) & 0xFFFFu)                  // tensor_dim0[31:16]
          | ((rows & 0xFFFFu) << 16);                      // tensor_dim1[15:0]
    g1[3] = ((rows >> 16) & 0xFFFFu)                       // tensor_dim1[31:16]
          | ((row_elems & 0xFFFFu) << 16);                 // tile_dim0
    g1[4] = rows & 0xFFFFu;                                // tile_dim1 (dim2=0)
    g1[5] = (unsigned)(row_stride_elems & 0xFFFFFFFFull);  // dim0_stride[31:0]
    g1[6] = (unsigned)((row_stride_elems >> 32) & 0xFFFFull);
    g1[7] = 0u;                                            // dim1_stride unused
    u32x4 gz = {0u, 0u, 0u, 0u};                           // groups 2/3 (2D)
    asm volatile("tensor_load_to_lds %0, %1, %2, %3"
                 :: "s"(g0), "s"(g1), "s"(gz), "s"(gz) : "memory");
}
__device__ __forceinline__ void wait_tensor() {
#if defined(__has_builtin)
#if __has_builtin(__builtin_amdgcn_s_wait_tensorcnt)
    __builtin_amdgcn_s_wait_tensorcnt(0);
    return;
#endif
#endif
    asm volatile("s_wait_tensorcnt 0x0" ::: "memory");
}

// ---- Fragment loaders (CDNA5 16-bit layouts, wave32) -----------------------
__device__ __forceinline__ v16bf load_a16(const bf16* p, int ld) {
    int lane = threadIdx.x & 31;
    const bf16* rp = p + (lane & 15) * ld + ((lane < 16) ? 0 : 16);
    v16bf a;
#pragma unroll
    for (int i = 0; i < 8; ++i) a[i] = rp[i];
#pragma unroll
    for (int i = 0; i < 8; ++i) a[8 + i] = rp[8 + i];
    return a;
}
__device__ __forceinline__ v16bf load_bT16(const bf16* p, int ld) {
    int lane = threadIdx.x & 31;
    const bf16* cp = p + (lane & 15) * ld + ((lane < 16) ? 0 : 16);
    v16bf b;
#pragma unroll
    for (int i = 0; i < 16; ++i) b[i] = cp[i];
    return b;
}

// ---------------------------------------------------------------------------
// Kernel 1: fold LoRA into weights:  W_eff[n][k] = W[n][k] + sum_r L2[n][r]*L1[r][k]
// ---------------------------------------------------------------------------
__global__ void prep_weights(const float* __restrict__ wq, const float* __restrict__ wk,
                             const float* __restrict__ wv, const float* __restrict__ wo,
                             const float* __restrict__ l1q, const float* __restrict__ l2q,
                             const float* __restrict__ l1k, const float* __restrict__ l2k,
                             const float* __restrict__ l1v, const float* __restrict__ l2v,
                             const float* __restrict__ l1o, const float* __restrict__ l2o,
                             bf16* __restrict__ out) {
    int idx = blockIdx.x * blockDim.x + threadIdx.x;
    if (idx >= 4 * D_ * D_) return;
    int mat = idx / (D_ * D_);
    int rem = idx % (D_ * D_);
    int n = rem / D_, k = rem % D_;
    const float *w, *l1, *l2;
    if      (mat == 0) { w = wq; l1 = l1q; l2 = l2q; }
    else if (mat == 1) { w = wk; l1 = l1k; l2 = l2k; }
    else if (mat == 2) { w = wv; l1 = l1v; l2 = l2v; }
    else               { w = wo; l1 = l1o; l2 = l2o; }
    float acc = w[n * D_ + k];
#pragma unroll
    for (int r = 0; r < R_; ++r) acc += l2[n * R_ + r] * l1[r * D_ + k];
    out[idx] = f2bf(acc);
}

// ---------------------------------------------------------------------------
// Kernel 2: f32 -> bf16 convert
// ---------------------------------------------------------------------------
__global__ void f32_to_bf16(const float* __restrict__ in, bf16* __restrict__ out, int n) {
    int idx = blockIdx.x * blockDim.x + threadIdx.x;
    if (idx < n) out[idx] = f2bf(in[idx]);
}

// ---------------------------------------------------------------------------
// Kernel 3: GEMM  C[m][n] = sum_k A[m][k] * W[n][k] (+ bias[n])
// Block tile 256(M) x 128(N), K-step 32. 8 waves, wave = 64x64 = 16 wmma.
// A tile staged by the Tensor Data Mover (one tensor_load_to_lds per k-step),
// B tile by per-lane async global->LDS copies.
// ---------------------------------------------------------------------------
template <bool OUT_F32>
__global__ __launch_bounds__(256) void gemm_bf16(
    const bf16* __restrict__ A, const bf16* __restrict__ W,
    const float* __restrict__ bias,
    bf16* __restrict__ outb, float* __restrict__ outf,
    int M, int N, int K) {
    const int LDA = 40;                           // 32 + 8 pad (bf16 elems)
    __shared__ __align__(16) bf16 As[256 * LDA];  // 20 KB
    __shared__ __align__(16) bf16 Bs[128 * LDA];  // 10 KB

    int tid  = threadIdx.x;
    int lane = tid & 31;
    int wid  = tid >> 5;
    int wm = wid >> 1, wn = wid & 1;              // 4x2 wave grid
    int m0 = blockIdx.x * 256;
    int n0 = blockIdx.y * 128;

    v8f acc[4][4];
#pragma unroll
    for (int i = 0; i < 4; ++i)
#pragma unroll
        for (int j = 0; j < 4; ++j)
#pragma unroll
            for (int r = 0; r < 8; ++r) acc[i][j][r] = 0.f;

    int lrow = tid >> 2;                          // 0..63
    int lcol = (tid & 3) * 8;                     // 0,8,16,24

    for (int kt = 0; kt < K; kt += 32) {
        // A tile (256x32) via TDM: pad 4 DW every 16 DW -> LDA=40 elems
        if (wid == 0)
            tdm_load_2d(As, A + (size_t)m0 * K + kt,
                        /*row_elems=*/32, /*rows=*/256,
                        /*stride=*/(unsigned long long)K,
                        /*pad_interval(16DW)=*/3u, /*pad_amount(4DW)=*/3u);
        // B tile (128x32) via per-lane async copies
#pragma unroll
        for (int j = 0; j < 2; ++j)
            async_copy_b128(&Bs[(lrow + 64 * j) * LDA + lcol],
                            W + (size_t)(n0 + lrow + 64 * j) * K + kt + lcol);
        wait_async();
        if (wid == 0) wait_tensor();
        __syncthreads();

        v16bf af[4], bf_[4];
#pragma unroll
        for (int mi = 0; mi < 4; ++mi)
            af[mi] = load_a16(&As[(wm * 64 + mi * 16) * LDA], LDA);
#pragma unroll
        for (int ni = 0; ni < 4; ++ni)
            bf_[ni] = load_bT16(&Bs[(wn * 64 + ni * 16) * LDA], LDA);
#pragma unroll
        for (int mi = 0; mi < 4; ++mi)
#pragma unroll
            for (int ni = 0; ni < 4; ++ni)
                acc[mi][ni] = wmma_bf16(af[mi], bf_[ni], acc[mi][ni]);
        __syncthreads();
    }

    // epilogue: D layout = rows striped across 16-lane halves
    int coll   = lane & 15;
    int rowsel = (lane < 16) ? 0 : 8;
#pragma unroll
    for (int mi = 0; mi < 4; ++mi)
#pragma unroll
        for (int ni = 0; ni < 4; ++ni) {
            int cm = m0 + wm * 64 + mi * 16 + rowsel;
            int cn = n0 + wn * 64 + ni * 16 + coll;
            float bs = bias ? bias[cn] : 0.f;
#pragma unroll
            for (int r = 0; r < 8; ++r) {
                float v = acc[mi][ni][r] + bs;
                if constexpr (OUT_F32) outf[(size_t)(cm + r) * N + cn] = v;
                else                   outb[(size_t)(cm + r) * N + cn] = f2bf(v);
            }
        }
}

// ---------------------------------------------------------------------------
// Kernel 4: RoPE in place on bf16 Q and K, layout (B,S,H,HD)
// ---------------------------------------------------------------------------
__global__ void rope_inplace(bf16* __restrict__ q, bf16* __restrict__ k,
                             const float* __restrict__ fcos, const float* __restrict__ fsin) {
    const int PH = HD_ / 2;
    int idx = blockIdx.x * blockDim.x + threadIdx.x;   // over B*S*H*PH
    if (idx >= B_ * S_ * H_ * PH) return;
    int p = idx % PH;
    int s = (idx / (PH * H_)) % S_;
    size_t base = (size_t)(idx / PH) * HD_ + 2 * p;
    float c  = fcos[s * PH + p];
    float sn = fsin[s * PH + p];
    float qe = bf2f(q[base]), qo = bf2f(q[base + 1]);
    q[base]     = f2bf(qe * c - qo * sn);
    q[base + 1] = f2bf(qe * sn + qo * c);
    float ke = bf2f(k[base]), ko = bf2f(k[base + 1]);
    k[base]     = f2bf(ke * c - ko * sn);
    k[base + 1] = f2bf(ke * sn + ko * c);
}

// ---------------------------------------------------------------------------
// Kernel 5: flash attention (causal, online softmax), bf16 in/out, f32 accum.
// Grid: (S/64, B*H). Block: 128 threads = 4 waves; wave w owns 16 query rows.
// K tile DMA'd by the TDM; V tile transposed at LDS-store time so every WMMA
// fragment read is a contiguous b128 LDS load.
// ---------------------------------------------------------------------------
__global__ __launch_bounds__(128) void flash_attn(
    const bf16* __restrict__ Q, const bf16* __restrict__ Kg,
    const bf16* __restrict__ Vg, bf16* __restrict__ O) {
    const int LD = 72;                               // 64 + 8 pad
    __shared__ __align__(16) bf16 Kt[64 * LD];       // K[key][hd]
    __shared__ __align__(16) bf16 VtT[64 * LD];      // V^T[hd][key]
    __shared__ __align__(16) bf16 Pt[4 * 16 * LD];   // per-wave P staging

    int tid  = threadIdx.x;
    int lane = tid & 31;
    int wid  = tid >> 5;
    int qblk = blockIdx.x;                           // S/64 tiles
    int bh   = blockIdx.y;                           // B*H
    int b = bh / H_, h = bh % H_;
    int q0    = qblk * 64;
    int qrow0 = q0 + wid * 16;
    const int RS = H_ * HD_;                         // row stride in (B,S,H,HD)

    // Q A-fragments held in registers for the whole KV sweep
    size_t qbase = (((size_t)b * S_ + qrow0) * H_ + h) * HD_;
    v16bf aq0 = load_a16(Q + qbase, RS);             // hd 0..31
    v16bf aq1 = load_a16(Q + qbase + 32, RS);        // hd 32..63

    float mprev[8], lsum[8];
    v8f oacc[4];
#pragma unroll
    for (int r = 0; r < 8; ++r) { mprev[r] = -3.0e38f; lsum[r] = 0.f; }
#pragma unroll
    for (int nt = 0; nt < 4; ++nt)
#pragma unroll
        for (int r = 0; r < 8; ++r) oacc[nt][r] = 0.f;

    const float scale = 0.125f;                      // 1/sqrt(64)
    int coll   = lane & 15;
    int rowsel = (lane < 16) ? 0 : 8;

    for (int kb = 0; kb <= qblk; ++kb) {
        // --- K tile via TDM: 64x64 bf16, row stride RS, pad 4 DW / 32 DW ----
        if (wid == 0)
            tdm_load_2d(Kt, Kg + (((size_t)b * S_ + kb * 64) * H_ + h) * HD_,
                        /*row_elems=*/64, /*rows=*/64,
                        /*stride=*/(unsigned long long)RS,
                        /*pad_interval(32DW)=*/4u, /*pad_amount(4DW)=*/3u);
        // --- V tile: transpose at store time: VtT[hd][key], b32 key-pairs ---
#pragma unroll
        for (int i = 0; i < 2; ++i) {
            int u  = tid + i * 128;                  // 0..255 = 32 keypairs x 8 chunks
            int kp = u >> 3;                         // key pair 0..31
            int hc = (u & 7) * 8;                    // hd chunk base
            size_t g0 = (((size_t)b * S_ + kb * 64 + 2 * kp) * H_ + h) * HD_ + hc;
            uint4 va = *(const uint4*)(Vg + g0);
            uint4 vb = *(const uint4*)(Vg + g0 + RS);
            const unsigned short* pa = (const unsigned short*)&va;
            const unsigned short* pb = (const unsigned short*)&vb;
#pragma unroll
            for (int j = 0; j < 8; ++j) {
                unsigned pk = (unsigned)pa[j] | ((unsigned)pb[j] << 16);
                *(unsigned*)(&VtT[(hc + j) * LD + 2 * kp]) = pk;
            }
        }
        if (wid == 0) wait_tensor();
        __syncthreads();

        // --- scores S = Q K^T for this wave's 16 rows x 64 keys -------------
        v8f sacc[4];
#pragma unroll
        for (int nt = 0; nt < 4; ++nt) {
            v8f c;
#pragma unroll
            for (int r = 0; r < 8; ++r) c[r] = 0.f;
            v16bf bk0 = load_bT16(&Kt[(nt * 16) * LD], LD);       // hd 0..31
            v16bf bk1 = load_bT16(&Kt[(nt * 16) * LD + 32], LD);  // hd 32..63
            c = wmma_bf16(aq0, bk0, c);
            c = wmma_bf16(aq1, bk1, c);
            sacc[nt] = c;
        }

        // --- scale + causal mask + row max (reduce over 16-lane half) -------
        bool diag = (kb == qblk);
        float rmax[8];
#pragma unroll
        for (int r = 0; r < 8; ++r) {
            float mx = -3.0e38f;
#pragma unroll
            for (int nt = 0; nt < 4; ++nt) {
                float v = sacc[nt][r] * scale;
                if (diag) {
                    int key = kb * 64 + nt * 16 + coll;
                    int qq  = qrow0 + rowsel + r;
                    if (key > qq) v = -3.0e38f;
                }
                sacc[nt][r] = v;
                mx = fmaxf(mx, v);
            }
#pragma unroll
            for (int m = 1; m < 16; m <<= 1) mx = fmaxf(mx, __shfl_xor(mx, m, 32));
            rmax[r] = mx;
        }

        // --- online softmax update ------------------------------------------
#pragma unroll
        for (int r = 0; r < 8; ++r) {
            float mnew = fmaxf(mprev[r], rmax[r]);
            float corr = __expf(mprev[r] - mnew);
            float rs = 0.f;
#pragma unroll
            for (int nt = 0; nt < 4; ++nt) {
                float p = __expf(sacc[nt][r] - mnew);
                sacc[nt][r] = p;
                rs += p;
            }
#pragma unroll
            for (int m = 1; m < 16; m <<= 1) rs += __shfl_xor(rs, m, 32);
            lsum[r] = lsum[r] * corr + rs;
            mprev[r] = mnew;
#pragma unroll
            for (int nt = 0; nt < 4; ++nt) oacc[nt][r] *= corr;
        }

        // --- stage P (16x64) in per-wave LDS region as bf16 -----------------
        bf16* pw = &Pt[wid * 16 * LD];
#pragma unroll
        for (int nt = 0; nt < 4; ++nt)
#pragma unroll
            for (int r = 0; r < 8; ++r)
                pw[(rowsel + r) * LD + nt * 16 + coll] = f2bf(sacc[nt][r]);

        // --- O += P @ V  (B-frags from VtT are contiguous b128 reads) -------
#pragma unroll
        for (int kk = 0; kk < 2; ++kk) {
            v16bf pa = load_a16(pw + kk * 32, LD);
#pragma unroll
            for (int nt = 0; nt < 4; ++nt) {
                v16bf bv = load_bT16(&VtT[(nt * 16) * LD + kk * 32], LD);
                oacc[nt] = wmma_bf16(pa, bv, oacc[nt]);
            }
        }
        __syncthreads();                              // protect K/V tiles
    }

    // --- normalize and write O (bf16, layout B,S,H,HD) ----------------------
    size_t obase = (((size_t)b * S_ + qrow0) * H_ + h) * HD_;
#pragma unroll
    for (int nt = 0; nt < 4; ++nt)
#pragma unroll
        for (int r = 0; r < 8; ++r) {
            float v = oacc[nt][r] / lsum[r];
            O[obase + (size_t)(rowsel + r) * RS + nt * 16 + coll] = f2bf(v);
        }
}

// ---------------------------------------------------------------------------
// Host launcher
// ---------------------------------------------------------------------------
extern "C" void kernel_launch(void* const* d_in, const int* in_sizes, int n_in,
                              void* d_out, int out_size, void* d_ws, size_t ws_size,
                              hipStream_t stream) {
    (void)in_sizes; (void)n_in; (void)out_size; (void)ws_size;
    const float* x    = (const float*)d_in[0];
    const float* fcos = (const float*)d_in[1];
    const float* fsin = (const float*)d_in[2];
    // d_in[3] = mask: causality computed analytically, never read
    const float* wq_w = (const float*)d_in[4];
    const float* wq_b = (const float*)d_in[5];
    const float* wk_w = (const float*)d_in[6];
    const float* wv_w = (const float*)d_in[7];
    const float* wo_w = (const float*)d_in[8];
    const float* wo_b = (const float*)d_in[9];
    const float* l1q = (const float*)d_in[10], *l2q = (const float*)d_in[11];
    const float* l1k = (const float*)d_in[12], *l2k = (const float*)d_in[13];
    const float* l1v = (const float*)d_in[14], *l2v = (const float*)d_in[15];
    const float* l1o = (const float*)d_in[16], *l2o = (const float*)d_in[17];
    float* out = (float*)d_out;

    // workspace layout (bf16)
    bf16* ws   = (bf16*)d_ws;
    size_t MD  = (size_t)MT_ * D_;        // 8192*1024
    size_t DD  = (size_t)D_ * D_;
    bf16* xb   = ws;                      // MD
    bf16* wqe  = xb + MD;                 // 4 * DD (q,k,v,o contiguous)
    bf16* wke  = wqe + DD;
    bf16* wve  = wke + DD;
    bf16* woe  = wve + DD;
    bf16* qb   = woe + DD;                // MD  (B,S,H,HD)
    bf16* kbuf = qb + MD;                 // MD
    bf16* vbuf = kbuf + MD;               // MD
    bf16* ab   = vbuf + MD;               // MD  attention output

    // 1. LoRA folded into weights, bf16
    int wtot = 4 * D_ * D_;
    prep_weights<<<(wtot + 255) / 256, 256, 0, stream>>>(
        wq_w, wk_w, wv_w, wo_w, l1q, l2q, l1k, l2k, l1v, l2v, l1o, l2o, wqe);

    // 2. x -> bf16
    f32_to_bf16<<<(int)((MD + 255) / 256), 256, 0, stream>>>(x, xb, (int)MD);

    // 3. Q/K/V projections (WMMA GEMMs)
    dim3 gg(MT_ / 256, D_ / 128);
    gemm_bf16<false><<<gg, 256, 0, stream>>>(xb, wqe, wq_b, qb, nullptr, MT_, D_, D_);
    gemm_bf16<false><<<gg, 256, 0, stream>>>(xb, wke, nullptr, kbuf, nullptr, MT_, D_, D_);
    gemm_bf16<false><<<gg, 256, 0, stream>>>(xb, wve, nullptr, vbuf, nullptr, MT_, D_, D_);

    // 4. RoPE on Q, K (in place)
    int rn = B_ * S_ * H_ * (HD_ / 2);
    rope_inplace<<<(rn + 255) / 256, 256, 0, stream>>>(qb, kbuf, fcos, fsin);

    // 5. causal flash attention
    dim3 ga(S_ / 64, B_ * H_);
    flash_attn<<<ga, 128, 0, stream>>>(qb, kbuf, vbuf, ab);

    // 6. output projection -> f32 result
    gemm_bf16<true><<<gg, 256, 0, stream>>>(ab, woe, wo_b, nullptr, out, MT_, D_, D_);
}
